// HeliosModel_64175401337192
// MI455X (gfx1250) — compile-verified, matrix operand order checked by reading, not run on hardware
//
#include <hip/hip_runtime.h>
#include <math.h>

// ---------------- model constants ----------------
constexpr int kB    = 2048;
constexpr int kMA   = 6;
constexpr int kS    = 11;
constexpr int kSP   = 10;
constexpr int kD    = 256;
constexpr int kH    = 4;
constexpr int kDH   = 64;
constexpr int kL    = 12;
constexpr int kG    = 2;
constexpr int kNREL = 500;
constexpr int kNTYPE= 1000;
constexpr int kNODE = 100000;
constexpr int kVOC  = kNODE - kNTYPE;
constexpr int kNCLS = kNREL + kNTYPE;
constexpr int kNG   = kB * kMA;          // 12288 GAT groups
constexpr float BIGF = 1000000.0f;

typedef __attribute__((ext_vector_type(16))) _Float16 v16h;
typedef __attribute__((ext_vector_type(8)))  _Float16 v8h;
typedef __attribute__((ext_vector_type(8)))  float    v8f;
typedef __attribute__((ext_vector_type(4)))  float    v4f;

// ---------------- device helpers ----------------
__device__ __forceinline__ float gelu_f(float x) {
  return 0.5f * x * (1.0f + erff(x * 0.70710678118654752f));
}

__device__ __forceinline__ float blockSum256(float v) {
  __shared__ float sm[256];
  int t = threadIdx.x;
  sm[t] = v; __syncthreads();
  for (int s = 128; s > 0; s >>= 1) { if (t < s) sm[t] += sm[t + s]; __syncthreads(); }
  float r = sm[0]; __syncthreads();
  return r;
}

__device__ __forceinline__ float blockMax256(float v) {
  __shared__ float sm[256];
  int t = threadIdx.x;
  sm[t] = v; __syncthreads();
  for (int s = 128; s > 0; s >>= 1) { if (t < s) sm[t] = fmaxf(sm[t], sm[t + s]); __syncthreads(); }
  float r = sm[0]; __syncthreads();
  return r;
}

// ---------------- WMMA GEMM: C = act(A * B + bias) ----------------
// A: (M,K) f32 row-major lda. B: (K,N) f32 row-major ldb, or transB: (N,K) ldb.
// Block = 128 threads (4 waves). Block tile: 64(M) x 32(N). K-step 32.
// B tile is staged in LDS as f16 [n][k] and shared by all 4 waves; each wave
// runs two v_wmma_f32_16x16x32_f16 per K-step (16x32 C tile per wave).
// Out-of-range rows/cols use clamped addresses; their products only reach
// store-guarded C entries, so no predicated loads are needed.
__global__ __launch_bounds__(128)
void wmma_gemm_kernel(const float* __restrict__ A, const float* __restrict__ Bm,
                      const float* __restrict__ bias, float* __restrict__ C,
                      int M, int N, int K, int lda, int ldb, int ldc,
                      int transB, int act)
{
  __shared__ _Float16 Bs[32][40];        // [n][k], 80B row pitch (16B-aligned, conflict-free)

  const int tid  = threadIdx.x;
  const int wave = tid >> 5;
  const int l    = tid & 31;
  const int lm   = l & 15;
  const int ko   = (l >> 4) * 8;         // lane-half K window
  const int tM   = blockIdx.y * 64 + wave * 16;
  const int tN   = blockIdx.x * 32;

  const int mr   = tM + lm;
  const float* Ar = A + (size_t)(mr < M ? mr : (M - 1)) * lda;

  // cooperative B-fill mapping: thread -> (n = tid&31, k = (tid>>5)*8 .. +7)
  const int fn  = tid & 31;
  const int fk  = (tid >> 5) * 8;
  const int gn  = tN + fn;
  const int gnc = (gn < N) ? gn : (N - 1);

  const int n0 = tN + lm, n1 = n0 + 16;
  const bool ok0 = (n0 < N), ok1 = (n1 < N);

  v8f acc0 = {}, acc1 = {};

  for (int kb = 0; kb < K; kb += 32) {
    // ---- cooperative B tile -> LDS (f16, transposed) ----
    float tmp[8];
    if (!transB) {
      const float* Bp = Bm + (size_t)(kb + fk) * ldb + gnc;
      #pragma unroll
      for (int i = 0; i < 8; ++i) tmp[i] = Bp[(size_t)i * ldb];
    } else {
      const float* Bp = Bm + (size_t)gnc * ldb + kb + fk;
      v4f x0 = *(const v4f*)(Bp);
      v4f x1 = *(const v4f*)(Bp + 4);
      tmp[0] = x0.x; tmp[1] = x0.y; tmp[2] = x0.z; tmp[3] = x0.w;
      tmp[4] = x1.x; tmp[5] = x1.y; tmp[6] = x1.z; tmp[7] = x1.w;
    }
    #pragma unroll
    for (int i = 0; i < 8; ++i) Bs[fn][fk + i] = (_Float16)tmp[i];
    __syncthreads();

    // ---- A fragment: 4x global_load_b128 + cvt ----
    if (kb + 64 < K) __builtin_prefetch(Ar + kb + 64, 0, 1);
    v16h af;
    {
      v4f a0 = *(const v4f*)(Ar + kb + ko);
      v4f a1 = *(const v4f*)(Ar + kb + ko + 4);
      v4f a2 = *(const v4f*)(Ar + kb + 16 + ko);
      v4f a3 = *(const v4f*)(Ar + kb + 16 + ko + 4);
      af[0] = (_Float16)a0.x; af[1] = (_Float16)a0.y; af[2]  = (_Float16)a0.z; af[3]  = (_Float16)a0.w;
      af[4] = (_Float16)a1.x; af[5] = (_Float16)a1.y; af[6]  = (_Float16)a1.z; af[7]  = (_Float16)a1.w;
      af[8] = (_Float16)a2.x; af[9] = (_Float16)a2.y; af[10] = (_Float16)a2.z; af[11] = (_Float16)a2.w;
      af[12]= (_Float16)a3.x; af[13]= (_Float16)a3.y; af[14] = (_Float16)a3.z; af[15] = (_Float16)a3.w;
    }

    // ---- B fragments: 16B ds loads from LDS ----
    v8h p00 = *(const v8h*)(&Bs[lm][ko]);
    v8h p01 = *(const v8h*)(&Bs[lm][16 + ko]);
    v8h p10 = *(const v8h*)(&Bs[lm + 16][ko]);
    v8h p11 = *(const v8h*)(&Bs[lm + 16][16 + ko]);
    v16h b0f = __builtin_shufflevector(p00, p01, 0,1,2,3,4,5,6,7,8,9,10,11,12,13,14,15);
    v16h b1f = __builtin_shufflevector(p10, p11, 0,1,2,3,4,5,6,7,8,9,10,11,12,13,14,15);

    acc0 = __builtin_amdgcn_wmma_f32_16x16x32_f16(false, af, false, b0f, (short)0, acc0, false, false);
    acc1 = __builtin_amdgcn_wmma_f32_16x16x32_f16(false, af, false, b1f, (short)0, acc1, false, false);
    __syncthreads();
  }

  // ---- epilogue: bias + optional exact GELU, guarded stores ----
  const int mo = (l >> 4) * 8;
  const float bv0 = (bias && ok0) ? bias[n0] : 0.f;
  const float bv1 = (bias && ok1) ? bias[n1] : 0.f;
  #pragma unroll
  for (int r = 0; r < 8; ++r) {
    int mc = tM + mo + r;
    if (mc < M) {
      if (ok0) { float v = acc0[r] + bv0; if (act) v = gelu_f(v); C[(size_t)mc * ldc + n0] = v; }
      if (ok1) { float v = acc1[r] + bv1; if (act) v = gelu_f(v); C[(size_t)mc * ldc + n1] = v; }
    }
  }
}

static void gemm(hipStream_t st, const float* A, const float* Bm, const float* bias, float* C,
                 int M, int N, int K, int lda, int ldb, int ldc, int transB, int act) {
  dim3 g((N + 31) / 32, (M + 63) / 64);
  wmma_gemm_kernel<<<g, 128, 0, st>>>(A, Bm, bias, C, M, N, K, lda, ldb, ldc, transB, act);
}

// ---------------- glue kernels ----------------
__global__ void embed_ln_kernel(const int* __restrict__ ids, const float* __restrict__ emb,
                                const float* __restrict__ g, const float* __restrict__ b,
                                float* __restrict__ out) {
  int row = blockIdx.x, d = threadIdx.x;
  float x = emb[(size_t)ids[row] * kD + d];
  float m = blockSum256(x) * (1.f / kD);
  float xm = x - m;
  float var = blockSum256(xm * xm) * (1.f / kD);
  out[(size_t)row * kD + d] = xm * rsqrtf(var + 1e-12f) * g[d] + b[d];
}

__global__ void itt_kernel(const int* __restrict__ input_ids, const int* __restrict__ ent_types,
                           int* __restrict__ itt, float* __restrict__ sgn1) {
  int idx = blockIdx.x * blockDim.x + threadIdx.x;
  if (idx >= kNG * kSP) return;
  int p = idx % kSP, gm = idx / kSP;
  int b = gm / kMA, m = gm % kMA;
  int id = input_ids[b * kS + 2 * m] - kNREL;
  int t = (id < 3) ? id : ent_types[(size_t)(id - 3) * (kSP + 1) + 1 + p];
  itt[idx] = t;
  sgn1[idx] = (t != 0) ? 1.f : 0.f;
}

__global__ void gather_rows_kernel(const int* __restrict__ idx, const float* __restrict__ emb,
                                   float* __restrict__ out) {
  int row = blockIdx.x, d = threadIdx.x;
  out[(size_t)row * kD + d] = emb[(size_t)idx[row] * kD + d];
}

__global__ void gat_st_kernel(const float* __restrict__ hW, const float* __restrict__ as,
                              const float* __restrict__ ad, float* __restrict__ sb,
                              float* __restrict__ tb) {
  int row = blockIdx.x, d = threadIdx.x;
  float x = hW[(size_t)row * kD + d];
  float s = blockSum256(x * as[d]);
  float t = blockSum256(x * ad[d]);
  if (d == 0) { sb[row] = s; tb[row] = t; }
}

__global__ void gat_attn_kernel(const float* __restrict__ sb, const float* __restrict__ tb,
                                const float* __restrict__ sgn, float* __restrict__ attn, int N) {
  int idx = blockIdx.x * blockDim.x + threadIdx.x;
  if (idx >= kNG * N) return;
  int g = idx / N;
  float si = sb[idx], sgi = sgn[idx];
  float e[16]; float mx = -1e30f;
  for (int j = 0; j < N; ++j) {
    float x = si + tb[g * N + j];
    x = (x > 0.f) ? x : 0.2f * x;
    x += BIGF * (sgi * sgn[g * N + j] - 1.f);
    e[j] = x; mx = fmaxf(mx, x);
  }
  float se = 0.f;
  for (int j = 0; j < N; ++j) { e[j] = expf(e[j] - mx); se += e[j]; }
  float inv = 1.f / se;
  for (int j = 0; j < N; ++j) attn[(size_t)idx * N + j] = e[j] * inv;
}

__global__ void gat_apply_kernel(const float* __restrict__ attn, const float* __restrict__ hW,
                                 float* __restrict__ hnew, int N, int total) {
  int idx = blockIdx.x * blockDim.x + threadIdx.x;
  if (idx >= total) return;
  int d = idx % kD, ri = idx / kD;
  int g = ri / N;
  const float* ar = attn + (size_t)ri * N;
  const float* hg = hW + (size_t)g * N * kD + d;
  float acc = 0.f;
  for (int j = 0; j < N; ++j) acc += ar[j] * hg[(size_t)j * kD];
  hnew[(size_t)ri * kD + d] = (acc > 0.f) ? acc : (expf(acc) - 1.f);
}

__global__ void gat_pool_kernel(const float* __restrict__ h, const float* __restrict__ sgn,
                                float* __restrict__ out, int N) {
  int idx = blockIdx.x * blockDim.x + threadIdx.x;
  if (idx >= kNG * kD) return;
  int d = idx % kD, g = idx / kD;
  float s = 0.f, den = 0.f;
  for (int j = 0; j < N; ++j) {
    float p = sgn[g * N + j];
    den += p;
    s += h[((size_t)g * N + j) * kD + d] * p;
  }
  out[(size_t)g * kD + d] = s / fmaxf(den, 1.f);
}

__global__ void h2_build_kernel(const int* __restrict__ itt, const float* __restrict__ node_emb,
                                const float* __restrict__ g1, const int* __restrict__ l2m,
                                float* __restrict__ h2) {
  int idx = blockIdx.x * blockDim.x + threadIdx.x;
  if (idx >= kNG * 15 * kD) return;
  int d = idx % kD, t = idx / kD;
  int i = t % 15, g = t / 15;
  float v;
  if (i < kSP) {
    v = node_emb[(size_t)itt[g * kSP + i] * kD + d];
  } else {
    int b = g / kMA, m = g % kMA;
    int src = l2m[m * (kMA - 1) + (i - kSP)];
    v = g1[((size_t)b * kMA + src) * kD + d];
  }
  h2[(size_t)t * kD + d] = v;
}

__global__ void sgn2_copy_kernel(const float* __restrict__ sgn1, float* __restrict__ sgn2) {
  int idx = blockIdx.x * blockDim.x + threadIdx.x;
  if (idx >= kNG * kSP) return;
  int g = idx / kSP, i = idx % kSP;
  sgn2[g * 15 + i] = sgn1[idx];
}

__global__ void sgn2_gather_kernel(const float* __restrict__ g1, const int* __restrict__ l2m,
                                   float* __restrict__ sgn2) {
  int gj = blockIdx.x;                     // g * 5 + j
  int g = gj / (kMA - 1), j = gj % (kMA - 1);
  int b = g / kMA, m = g % kMA;
  int src = l2m[m * (kMA - 1) + j];
  float x = g1[((size_t)b * kMA + src) * kD + threadIdx.x];
  float s = blockSum256(x);
  if (threadIdx.x == 0) sgn2[g * 15 + kSP + j] = (s != 0.f) ? 1.f : 0.f;
}

__global__ void scatter_even_kernel(const float* __restrict__ g2, float* __restrict__ emb) {
  int idx = blockIdx.x * blockDim.x + threadIdx.x;
  if (idx >= kNG * kD) return;
  int d = idx % kD, t = idx / kD;
  int b = t / kMA, m = t % kMA;
  emb[((size_t)b * kS + 2 * m) * kD + d] = g2[(size_t)t * kD + d];
}

__global__ void mask_set_kernel(const int* __restrict__ mask_pos, const int* __restrict__ mask_type,
                                const float* __restrict__ node_emb, float* __restrict__ emb) {
  int idx = blockIdx.x * blockDim.x + threadIdx.x;
  if (idx >= kB * kD) return;
  int d = idx % kD, b = idx / kD;
  int row = (mask_type[b] == 1) ? 2 : 1;
  emb[((size_t)b * kS + mask_pos[b]) * kD + d] = node_emb[(size_t)row * kD + d];
}

__global__ void ekev_kernel(const int* __restrict__ edge_labels, const float* __restrict__ ke,
                            const float* __restrict__ ve, float* __restrict__ ek,
                            float* __restrict__ ev) {
  int idx = blockIdx.x * blockDim.x + threadIdx.x;
  if (idx >= kB * kS * kS * kDH) return;
  int d = idx % kDH, t = idx / kDH;
  int lab = edge_labels[t];
  float sg = (lab != 0) ? 1.f : 0.f;
  ek[idx] = ke[lab * kDH + d] * sg;
  ev[idx] = ve[lab * kDH + d] * sg;
}

__global__ void scores_kernel(const float* __restrict__ qkv, const float* __restrict__ ek,
                              const float* __restrict__ mask, float* __restrict__ scr) {
  int idx = blockIdx.x * blockDim.x + threadIdx.x;
  if (idx >= kB * kH * kS * kS) return;
  int j = idx % kS, t = idx / kS;
  int i = t % kS; t /= kS;
  int h = t % kH, b = t / kH;
  const float* q  = qkv + ((size_t)b * kS + i) * 3 * kD + h * kDH;
  const float* kk = qkv + ((size_t)b * kS + j) * 3 * kD + kD + h * kDH;
  const float* ep = ek  + (((size_t)b * kS + i) * kS + j) * kDH;
  float s = 0.f;
  #pragma unroll 8
  for (int d = 0; d < kDH; ++d) s += q[d] * (kk[d] + ep[d]);   // q.k + q.ek
  s = s * 0.125f + BIGF * (mask[b * kS + i] * mask[b * kS + j] - 1.f);
  scr[idx] = s;
}

__global__ void attn_softmax_kernel(float* __restrict__ scr) {
  int idx = blockIdx.x * blockDim.x + threadIdx.x;
  if (idx >= kB * kH * kS) return;
  float* r = scr + (size_t)idx * kS;
  float mx = -1e30f;
  for (int j = 0; j < kS; ++j) mx = fmaxf(mx, r[j]);
  float se = 0.f;
  for (int j = 0; j < kS; ++j) { r[j] = expf(r[j] - mx); se += r[j]; }
  float inv = 1.f / se;
  for (int j = 0; j < kS; ++j) r[j] *= inv;
}

__global__ void ctx_kernel(const float* __restrict__ scr, const float* __restrict__ qkv,
                           const float* __restrict__ ev, float* __restrict__ ctx) {
  int idx = blockIdx.x * blockDim.x + threadIdx.x;
  if (idx >= kB * kS * kD) return;
  int d = idx % kDH, t = idx / kDH;
  int h = t % kH; t /= kH;
  int i = t % kS, b = t / kS;
  const float* at = scr + (((size_t)b * kH + h) * kS + i) * kS;
  float acc = 0.f;
  for (int j = 0; j < kS; ++j) {
    float v  = qkv[((size_t)b * kS + j) * 3 * kD + 2 * kD + h * kDH + d];
    float vv = ev[(((size_t)b * kS + i) * kS + j) * kDH + d];
    acc += at[j] * (v + vv);
  }
  ctx[((size_t)b * kS + i) * kD + h * kDH + d] = acc;
}

__global__ void add_ln_kernel(float* __restrict__ x, const float* __restrict__ y,
                              const float* __restrict__ g, const float* __restrict__ b, float eps) {
  int row = blockIdx.x, d = threadIdx.x;
  float v = x[(size_t)row * kD + d] + y[(size_t)row * kD + d];
  float m = blockSum256(v) * (1.f / kD);
  float xm = v - m;
  float var = blockSum256(xm * xm) * (1.f / kD);
  x[(size_t)row * kD + d] = xm * rsqrtf(var + eps) * g[d] + b[d];
}

__global__ void hm_kernel(const float* __restrict__ x, const int* __restrict__ mask_pos,
                          const float* __restrict__ g, const float* __restrict__ b,
                          float* __restrict__ hm) {
  int bi = blockIdx.x, d = threadIdx.x;
  float v = x[((size_t)bi * kS + mask_pos[bi]) * kD + d];
  v = gelu_f(v);
  float m = blockSum256(v) * (1.f / kD);
  float xm = v - m;
  float var = blockSum256(xm * xm) * (1.f / kD);
  hm[(size_t)bi * kD + d] = xm * rsqrtf(var + 1e-7f) * g[d] + b[d];
}

__global__ void zero4_kernel(float* __restrict__ a) {
  if (threadIdx.x < 4) a[threadIdx.x] = 0.f;
}

__global__ void loss_kernel(const float* __restrict__ fc, const float* __restrict__ gt,
                            const int* __restrict__ mask_type, float* __restrict__ accv) {
  int b = blockIdx.x, tid = threadIdx.x;
  const float* f2 = fc + (size_t)b * kNCLS;
  const float* f1 = f2 + kNREL;
  const float* gr = gt + (size_t)b * kNCLS;
  const float* gtp = gr + kNREL;
  // relation head (500)
  float mx = -1e30f;
  for (int c = tid; c < kNREL; c += 256) mx = fmaxf(mx, f2[c]);
  mx = blockMax256(mx);
  float se = 0.f, dt = 0.f, gs = 0.f;
  for (int c = tid; c < kNREL; c += 256) { se += expf(f2[c] - mx); dt += gr[c] * f2[c]; gs += gr[c]; }
  se = blockSum256(se); dt = blockSum256(dt); gs = blockSum256(gs);
  float ce_r = (mx + logf(se)) - dt / gs;
  // type head (1000)
  float mx1 = -1e30f;
  for (int c = tid; c < kNTYPE; c += 256) mx1 = fmaxf(mx1, f1[c]);
  mx1 = blockMax256(mx1);
  float se1 = 0.f, dt1 = 0.f, gs1 = 0.f;
  for (int c = tid; c < kNTYPE; c += 256) { se1 += expf(f1[c] - mx1); dt1 += gtp[c] * f1[c]; gs1 += gtp[c]; }
  se1 = blockSum256(se1); dt1 = blockSum256(dt1); gs1 = blockSum256(gs1);
  float ce_t = (mx1 + logf(se1)) - dt1 / gs1;
  if (tid == 0) {
    int mt = mask_type[b];
    if (mt == -1) { atomicAdd(&accv[0], ce_r); atomicAdd(&accv[1], 1.f); }
    else if (mt == 1) { atomicAdd(&accv[2], ce_t); atomicAdd(&accv[3], 1.f); }
  }
}

__global__ void fc_mask_kernel(const int* __restrict__ mask_type, float* __restrict__ fc) {
  int idx = blockIdx.x * blockDim.x + threadIdx.x;
  if (idx >= kB * kNCLS) return;
  int c = idx % kNCLS, b = idx / kNCLS;
  float ind = ((c < kNREL) ? -1.f : 1.f) * (float)mask_type[b];
  fc[idx] += BIGF * (((ind > 0.f) ? ind : 0.f) - 1.f);
}

__global__ void finalize_kernel(const float* __restrict__ accv, float* __restrict__ out) {
  out[0] = accv[0] / fmaxf(accv[1], 1.f) + accv[2] / fmaxf(accv[3], 1.f);
}

// ---------------- host orchestration ----------------
extern "C" void kernel_launch(void* const* d_in, const int* in_sizes, int n_in,
                              void* d_out, int out_size, void* d_ws, size_t ws_size,
                              hipStream_t stream)
{
  (void)in_sizes; (void)n_in; (void)out_size; (void)ws_size;
  const int*   input_ids   = (const int*)  d_in[0];
  const float* input_mask  = (const float*)d_in[1];
  const int*   edge_labels = (const int*)  d_in[2];
  const int*   mask_pos    = (const int*)  d_in[3];
  /* d_in[4] mask_label unused by forward */
  const int*   mask_type   = (const int*)  d_in[5];
  const int*   ent_types   = (const int*)  d_in[6];
  const float* gt          = (const float*)d_in[7];
  const int*   l2m         = (const int*)  d_in[8];
  const float* node_emb    = (const float*)d_in[9];
  const float* ln1_g = (const float*)d_in[10];
  const float* ln1_b = (const float*)d_in[11];
  const float* ln2_g = (const float*)d_in[12];
  const float* ln2_b = (const float*)d_in[13];
  const float* fc2_bias = (const float*)d_in[14];
  const float* fc3_bias = (const float*)d_in[15];
  const float* edge_k = (const float*)d_in[16];
  const float* edge_v = (const float*)d_in[17];
  const float* gat1_W  = (const float*)d_in[18];
  const float* gat1_as = (const float*)d_in[19];
  const float* gat1_ad = (const float*)d_in[20];
  const float* gat2_W  = (const float*)d_in[21];
  const float* gat2_as = (const float*)d_in[22];
  const float* gat2_ad = (const float*)d_in[23];
  const float* enc_wqkv = (const float*)d_in[24];
  const float* enc_wo   = (const float*)d_in[25];
  const float* enc_l1g  = (const float*)d_in[26];
  const float* enc_l1b  = (const float*)d_in[27];
  const float* enc_w1   = (const float*)d_in[28];
  const float* enc_b1   = (const float*)d_in[29];
  const float* enc_w2   = (const float*)d_in[30];
  const float* enc_b2   = (const float*)d_in[31];
  const float* enc_l2g  = (const float*)d_in[32];
  const float* enc_l2b  = (const float*)d_in[33];

  float* out   = (float*)d_out;
  float* fcout = out + 1;                       // (B, NCLS) row-major

  // ---- workspace bump allocator (floats) ----
  float* W = (float*)d_ws;
  size_t off = 0;
  auto take = [&](size_t n) -> float* {
    float* p = W + off; off += (n + 255) & ~(size_t)255; return p;
  };
  float* emb  = take((size_t)kB * kS * kD);
  float* g1   = take((size_t)kNG * kD);
  float* g2b  = take((size_t)kNG * kD);
  int*   itt  = (int*)take((size_t)kNG * kSP);
  float* sgn1 = take((size_t)kNG * kSP);
  float* sgn2 = take((size_t)kNG * 15);
  float* hm   = take((size_t)kB * kD);
  float* accv = take(256);
  size_t bigBase = off;
  float* hbuf = take((size_t)kNG * 15 * kD);
  float* hWb  = take((size_t)kNG * 15 * kD);
  float* hnew = take((size_t)kNG * 15 * kD);
  float* sb   = take((size_t)kNG * 15);
  float* tb   = take((size_t)kNG * 15);
  float* attn = take((size_t)kNG * 15 * 15);
  // encoder buffers alias the (larger) GAT scratch region
  off = bigBase;
  float* qkv = take((size_t)kB * kS * 3 * kD);
  float* scr = take((size_t)kB * kH * kS * kS);
  float* ctx = take((size_t)kB * kS * kD);
  float* ffh = take((size_t)kB * kS * 512);
  float* ff2 = take((size_t)kB * kS * kD);
  float* ekb = take((size_t)kB * kS * kS * kDH);
  float* evb = take((size_t)kB * kS * kS * kDH);

  const int T = 256;
  auto nb = [&](long long n) { return (unsigned)((n + T - 1) / T); };

  // 1. token embedding + LN1
  embed_ln_kernel<<<kB * kS, T, 0, stream>>>(input_ids, node_emb, ln1_g, ln1_b, emb);

  // 2. type matrix gather + sign
  itt_kernel<<<nb((long long)kNG * kSP), T, 0, stream>>>(input_ids, ent_types, itt, sgn1);
  gather_rows_kernel<<<kNG * kSP, T, 0, stream>>>(itt, node_emb, hbuf);

  // 3. GAT stage 1 (N = 10)
  {
    float* hc = hbuf; float* hn = hnew;
    for (int gl = 0; gl < kG; ++gl) {
      gemm(stream, hc, gat1_W + (size_t)gl * kD * kD, nullptr, hWb,
           kNG * kSP, kD, kD, kD, kD, kD, 0, 0);
      gat_st_kernel<<<kNG * kSP, T, 0, stream>>>(hWb, gat1_as + gl * kD, gat1_ad + gl * kD, sb, tb);
      gat_attn_kernel<<<nb((long long)kNG * kSP), T, 0, stream>>>(sb, tb, sgn1, attn, kSP);
      gat_apply_kernel<<<nb((long long)kNG * kSP * kD), T, 0, stream>>>(attn, hWb, hn, kSP, kNG * kSP * kD);
      float* t2 = hc; hc = hn; hn = t2;
    }
    gat_pool_kernel<<<nb((long long)kNG * kD), T, 0, stream>>>(hc, sgn1, g1, kSP);
  }

  // 4. build h2 (N = 15) and sgn2
  h2_build_kernel<<<nb((long long)kNG * 15 * kD), T, 0, stream>>>(itt, node_emb, g1, l2m, hbuf);
  sgn2_copy_kernel<<<nb((long long)kNG * kSP), T, 0, stream>>>(sgn1, sgn2);
  sgn2_gather_kernel<<<kNG * (kMA - 1), T, 0, stream>>>(g1, l2m, sgn2);

  // 5. GAT stage 2 (N = 15)
  {
    float* hc = hbuf; float* hn = hnew;
    for (int gl = 0; gl < kG; ++gl) {
      gemm(stream, hc, gat2_W + (size_t)gl * kD * kD, nullptr, hWb,
           kNG * 15, kD, kD, kD, kD, kD, 0, 0);
      gat_st_kernel<<<kNG * 15, T, 0, stream>>>(hWb, gat2_as + gl * kD, gat2_ad + gl * kD, sb, tb);
      gat_attn_kernel<<<nb((long long)kNG * 15), T, 0, stream>>>(sb, tb, sgn2, attn, 15);
      gat_apply_kernel<<<nb((long long)kNG * 15 * kD), T, 0, stream>>>(attn, hWb, hn, 15, kNG * 15 * kD);
      float* t2 = hc; hc = hn; hn = t2;
    }
    gat_pool_kernel<<<nb((long long)kNG * kD), T, 0, stream>>>(hc, sgn2, g2b, 15);
  }

  // 6. scatter GAT output into even positions, then masked-token embedding
  scatter_even_kernel<<<nb((long long)kNG * kD), T, 0, stream>>>(g2b, emb);
  mask_set_kernel<<<nb((long long)kB * kD), T, 0, stream>>>(mask_pos, mask_type, node_emb, emb);

  // 7. edge K/V embeddings
  ekev_kernel<<<nb((long long)kB * kS * kS * kDH), T, 0, stream>>>(edge_labels, edge_k, edge_v, ekb, evb);

  // 8. transformer encoder (12 layers)
  for (int l = 0; l < kL; ++l) {
    const float* wqkv = enc_wqkv + (size_t)l * kD * 3 * kD;
    const float* wo   = enc_wo   + (size_t)l * kD * kD;
    const float* l1g  = enc_l1g  + (size_t)l * kD;
    const float* l1b  = enc_l1b  + (size_t)l * kD;
    const float* w1   = enc_w1   + (size_t)l * kD * 512;
    const float* b1   = enc_b1   + (size_t)l * 512;
    const float* w2   = enc_w2   + (size_t)l * 512 * kD;
    const float* b2   = enc_b2   + (size_t)l * kD;
    const float* l2g  = enc_l2g  + (size_t)l * kD;
    const float* l2b  = enc_l2b  + (size_t)l * kD;

    gemm(stream, emb, wqkv, nullptr, qkv, kB * kS, 3 * kD, kD, kD, 3 * kD, 3 * kD, 0, 0);
    scores_kernel<<<nb((long long)kB * kH * kS * kS), T, 0, stream>>>(qkv, ekb, input_mask, scr);
    attn_softmax_kernel<<<nb((long long)kB * kH * kS), T, 0, stream>>>(scr);
    ctx_kernel<<<nb((long long)kB * kS * kD), T, 0, stream>>>(scr, qkv, evb, ctx);
    gemm(stream, ctx, wo, nullptr, ff2, kB * kS, kD, kD, kD, kD, kD, 0, 0);
    add_ln_kernel<<<kB * kS, T, 0, stream>>>(emb, ff2, l1g, l1b, 1e-12f);
    gemm(stream, emb, w1, b1, ffh, kB * kS, 512, kD, kD, 512, 512, 0, /*gelu*/1);
    gemm(stream, ffh, w2, b2, ff2, kB * kS, kD, 512, 512, kD, kD, 0, 0);
    add_ln_kernel<<<kB * kS, T, 0, stream>>>(emb, ff2, l2g, l2b, 1e-12f);
  }

  // 9. masked-position head: gather -> gelu -> LN2
  hm_kernel<<<kB, T, 0, stream>>>(emb, mask_pos, ln2_g, ln2_b, hm);

  // 10. output heads: fc2 = hm @ node_emb[3:503]^T + b2 ; fc1 = hm @ node_emb[VOC:]^T + b3
  gemm(stream, hm, node_emb + (size_t)3 * kD,    fc2_bias, fcout,         kB, kNREL,  kD, kD, kD, kNCLS, 1, 0);
  gemm(stream, hm, node_emb + (size_t)kVOC * kD, fc3_bias, fcout + kNREL, kB, kNTYPE, kD, kD, kD, kNCLS, 1, 0);

  // 11. loss on pre-mask logits, then BIG masking of fc_out, then finalize
  zero4_kernel<<<1, 32, 0, stream>>>(accv);
  loss_kernel<<<kB, T, 0, stream>>>(fcout, gt, mask_type, accv);
  fc_mask_kernel<<<nb((long long)kB * kNCLS), T, 0, stream>>>(mask_type, fcout);
  finalize_kernel<<<1, 1, 0, stream>>>(accv, out);
}
